// TensorProductWrapper_41566693490835
// MI455X (gfx1250) — matrix operand order repaired and checked by reading, not compiled, not run on hardware
//
#include <hip/hip_runtime.h>
#include <hip/hip_bf16.h>

#define MULDIM   64
#define BATCH    8192
#define R_TILES  4            // 16-row b-tiles per workgroup (64 rows)
#define WG_THREADS 320        // 10 wave32 waves
#define TENSOR_ELEMS (MULDIM*MULDIM*MULDIM)   // 262144

// Dynamic LDS map (96 KiB):
//   [0      .. 32768)  x tiles: x0 @0 (8KB), x1[i] @ 8KB*(1+i)
//   [32768  .. 98304)  4 weight buffers, 16 KiB each (4 tensors x 2048 bf16)
//   epilogue reuses [0 .. 49152) for partial-sum tiles
#define WBUF0     32768u
#define WBUFSZ    16384u
#define NBUF      4
#define LDS_BYTES 98304

typedef __bf16 bf16;
typedef __attribute__((ext_vector_type(16))) __bf16 v16bf;
typedef __attribute__((ext_vector_type(8)))  float  v8f;

union Frag {
  v16bf v;
  unsigned int u[8];
  uint4 q[2];
};

// workspace layout (bytes)
#define WT_OFF 0                                   // 4 tensors, WMMA-B-tiled bf16: 2 MiB
#define X0_OFF (4*TENSOR_ELEMS*2)                  // x0 bf16 [b][u]: 1 MiB
#define X1_OFF (X0_OFF + BATCH*MULDIM*2)           // x1 bf16 [i][b][v]: 3 MiB

// ---------------------------------------------------------------------------
// Weight prep: f32 -> bf16, fold scales, swizzle into per-lane WMMA-B fragment
// order: [t][vhalf][u][ntile][lane][j]   (lane<16: K=j, lane>=16: K=16+j)
// ---------------------------------------------------------------------------
__global__ void prep_w(const float* __restrict__ w000, const float* __restrict__ w110,
                       const float* __restrict__ w011, const float* __restrict__ w101,
                       bf16* __restrict__ wt) {
  int idx = blockIdx.x * 256 + threadIdx.x;        // 4 * 64^3 threads
  int t = idx >> 18;
  int r = idx & (TENSOR_ELEMS - 1);
  int u = r >> 12;
  int v = (r >> 6) & 63;
  int w = r & 63;
  const float* src = (t == 0) ? w000 : (t == 1) ? w110 : (t == 2) ? w011 : w101;
  // ALPHA0 = sqrt(1/8192); ALPHA1*INV_SQRT3 == ALPHA0; w110 gets extra 1/sqrt(3)
  const float A0 = 0.011048543456039806f;
  const float INV3 = 0.5773502691896258f;
  float sc = (t == 1) ? A0 * INV3 : A0;
  float val = src[r] * sc;
  int vhalf = v >> 5, j = v & 15, lhi = (v >> 4) & 1;
  int nt = w >> 4, n = w & 15;
  int lane = lhi * 16 + n;
  wt[t * TENSOR_ELEMS + (vhalf * 64 + u) * 2048 + nt * 512 + lane * 16 + j] = (bf16)val;
}

// ---------------------------------------------------------------------------
// x prep: f32 -> bf16; de-interleave x1 (k stride-3) into [i][b][v]
// ---------------------------------------------------------------------------
__global__ void prep_x(const float* __restrict__ x,
                       bf16* __restrict__ x0bf, bf16* __restrict__ x1bf) {
  int idx = blockIdx.x * 256 + threadIdx.x;        // 8192 * 256 threads
  int b = idx >> 8, c = idx & 255;
  float val = x[idx];
  if (c < 64) {
    x0bf[b * 64 + c] = (bf16)val;
  } else {
    int cc = c - 64;
    int v = cc / 3, i = cc - 3 * v;
    x1bf[((i * BATCH + b) << 6) + v] = (bf16)val;
  }
}

// ---------------------------------------------------------------------------
// Async-stage one K=32 weight chunk (4 tensors x 2048 bf16 = 16 KiB) into LDS.
// Waves 0-7 only: exactly 4 GLOBAL_LOAD_ASYNC_TO_LDS_B128 per wave per chunk,
// so ASYNCcnt is wave-uniform and partial waits (s_wait_asynccnt N) are exact.
// ---------------------------------------------------------------------------
__device__ __forceinline__ void stage_async(const bf16* __restrict__ wt, int kc,
                                            unsigned int ldsbase, int tid) {
  const int vhalf = kc >> 6, u = kc & 63;
  const unsigned long long gbase = (unsigned long long)(uintptr_t)wt;
#pragma unroll
  for (int j = 0; j < 4; ++j) {
    int i = tid + j * 256;
    int t = i >> 8;
    unsigned int voff = (unsigned)t * 524288u +
                        (unsigned)(vhalf * 64 + u) * 4096u + (unsigned)(i & 255) * 16u;
    unsigned int lds = ldsbase + (unsigned)i * 16u;
    asm volatile("global_load_async_to_lds_b128 %0, %1, %2"
                 :: "v"(lds), "v"(voff), "s"(gbase) : "memory");
  }
}

// ---------------------------------------------------------------------------
// Main tensor-product kernel.
// 10 waves: w0 = p000 | w1-3 = p110_i | w4-6 = p011_k | w7-9 = p101_k
// Each wave: 4 b-tiles x 4 n-tiles f32 accumulators, K = 4096 per output.
// ---------------------------------------------------------------------------
__global__ __launch_bounds__(WG_THREADS, 1)
void tp_main(const bf16* __restrict__ wt, const bf16* __restrict__ x0bf,
             const bf16* __restrict__ x1bf, float* __restrict__ out) {
  extern __shared__ __align__(16) unsigned char smem[];
  const int tid  = threadIdx.x;
  const int wave = tid >> 5, lane = tid & 31;
  const int row  = lane & 15;               // A/C row within tile
  const int rofs = (lane >> 4) << 3;        // hi half-wave offset (M for C; A's K offset is 2*rofs)
  const int bbase = blockIdx.x * (R_TILES * 16);
  const unsigned int smbase = (unsigned int)(uintptr_t)smem;

  // ---- async-stage x chunk (64 rows, 32 KiB) into LDS (GV form: per-lane
  //      64-bit address, since the x1[i] region bases are lane-varying) ----
  if (wave < 8) {
#pragma unroll
    for (int j = 0; j < 8; ++j) {
      int i = tid + j * 256;
      int region = i >> 9, off = i & 511;
      const bf16* src = (region == 0) ? (x0bf + bbase * 64)
                                      : (x1bf + ((region - 1) * BATCH + bbase) * 64);
      unsigned long long gaddr = (unsigned long long)(uintptr_t)src +
                                 (unsigned long long)off * 16u;
      unsigned int lds = smbase + (unsigned)i * 16u;
      asm volatile("global_load_async_to_lds_b128 %0, %1, off"
                   :: "v"(lds), "v"(gaddr) : "memory");
    }
    // prologue: prefetch weight chunks 0..2 into buffers 0..2
    stage_async(wt, 0, smbase + WBUF0 + 0u * WBUFSZ, tid);
    stage_async(wt, 1, smbase + WBUF0 + 1u * WBUFSZ, tid);
    stage_async(wt, 2, smbase + WBUF0 + 2u * WBUFSZ, tid);
  }

  // per-wave role: which tensor, which LDS x arrays feed xu (scalar) / xv (vector)
  const int tensor = (wave == 0) ? 0 : (wave <= 3) ? 1 : (wave <= 6) ? 2 : 3;
  const int xu_off = (wave == 0) ? 0 : (wave <= 3) ? wave * 8192
                    : (wave <= 6) ? 0 : (wave - 6) * 8192;
  const int xv_off = (wave == 0) ? 0 : (wave <= 3) ? wave * 8192
                    : (wave <= 6) ? (wave - 3) * 8192 : 0;

  v8f acc[R_TILES][4];
#pragma unroll
  for (int bt = 0; bt < R_TILES; ++bt)
#pragma unroll
    for (int nt = 0; nt < 4; ++nt) { v8f z = {}; acc[bt][nt] = z; }

  Frag xv[R_TILES];
  for (int kc = 0; kc < 128; ++kc) {
    // wait: chunk kc complete; async loads finish in order, so <=8 leaves only
    // the two newer chunks (4 insts each) in flight.  Tail: fewer in flight.
    if (kc < 126)       asm volatile("s_wait_asynccnt 0x8" ::: "memory");
    else if (kc == 126) asm volatile("s_wait_asynccnt 0x4" ::: "memory");
    else                asm volatile("s_wait_asynccnt 0x0" ::: "memory");
    __syncthreads();   // chunk kc visible to all waves; all reads of kc-1 done

    // prefetch chunk kc+3 into buffer (kc+3)&3 (its last readers ran at kc-1)
    if (wave < 8 && kc < 125)
      stage_async(wt, kc + 3, smbase + WBUF0 + (unsigned)((kc + 3) & 3) * WBUFSZ, tid);

    if ((kc & 63) == 0) {
      // refresh resident xv fragments for this vhalf (A-layout K order)
      const int voff = (kc >> 6) * 32 + 2 * rofs;
#pragma unroll
      for (int bt = 0; bt < R_TILES; ++bt) {
        const unsigned char* p = smem + xv_off + (((bt * 16 + row) * 64 + voff) << 1);
        xv[bt].q[0] = *(const uint4*)p;
        xv[bt].q[1] = *(const uint4*)(p + 16);
      }
    }

    const int u = kc & 63;
    // A-gen: splat xu and multiply into resident xv (co-executes with WMMA)
    Frag a[R_TILES];
#pragma unroll
    for (int bt = 0; bt < R_TILES; ++bt) {
      unsigned int h =
          *(const unsigned short*)(smem + xu_off + (((bt * 16 + row) * 64 + u) << 1));
      unsigned int xu2 = h * 0x10001u;
#pragma unroll
      for (int g = 0; g < 8; ++g)
        asm("v_pk_mul_bf16 %0, %1, %2"
            : "=v"(a[bt].u[g]) : "v"(xu2), "v"(xv[bt].u[g]));
    }
#pragma unroll
    for (int nt = 0; nt < 4; ++nt) {
      Frag b;
      const unsigned char* p = smem + WBUF0 + (unsigned)(kc & 3) * WBUFSZ +
                               (unsigned)(tensor * 4096 + nt * 1024 + lane * 32);
      b.q[0] = *(const uint4*)p;
      b.q[1] = *(const uint4*)(p + 16);
#pragma unroll
      for (int bt = 0; bt < R_TILES; ++bt)
        acc[bt][nt] = __builtin_amdgcn_wmma_f32_16x16x32_bf16(
            false, a[bt].v, false, b.v, (short)0, acc[bt][nt], false, false);
    }
  }
  __syncthreads();   // protect LDS reuse below from in-flight compute reads

  // ---- epilogue: deterministic partial-sum combine via LDS, then store ----
  float* lp = (float*)smem;
  // phase 1: p110 partials -> wave 0 adds to p000 -> out0
  if (wave >= 1 && wave <= 3) {
    const int src = wave - 1;
#pragma unroll
    for (int bt = 0; bt < R_TILES; ++bt)
#pragma unroll
      for (int nt = 0; nt < 4; ++nt) {
        float* tp = lp + ((src * R_TILES + bt) * 4 + nt) * 256;
#pragma unroll
        for (int r = 0; r < 8; ++r) tp[(r + rofs) * 16 + row] = acc[bt][nt][r];
      }
  }
  __syncthreads();
  if (wave == 0) {
#pragma unroll
    for (int bt = 0; bt < R_TILES; ++bt)
#pragma unroll
      for (int nt = 0; nt < 4; ++nt)
#pragma unroll
        for (int r = 0; r < 8; ++r) {
          float s = acc[bt][nt][r];
#pragma unroll
          for (int src = 0; src < 3; ++src)
            s += lp[((src * R_TILES + bt) * 4 + nt) * 256 + (r + rofs) * 16 + row];
          out[(bbase + bt * 16 + r + rofs) * 256 + nt * 16 + row] = s;
        }
  }
  __syncthreads();
  // phase 2: p101 partials -> waves 4..6 combine with p011 -> out1 (stride-3)
  if (wave >= 7) {
    const int src = wave - 7;
#pragma unroll
    for (int bt = 0; bt < R_TILES; ++bt)
#pragma unroll
      for (int nt = 0; nt < 4; ++nt) {
        float* tp = lp + ((src * R_TILES + bt) * 4 + nt) * 256;
#pragma unroll
        for (int r = 0; r < 8; ++r) tp[(r + rofs) * 16 + row] = acc[bt][nt][r];
      }
  }
  __syncthreads();
  if (wave >= 4 && wave <= 6) {
    const int k = wave - 4;
#pragma unroll
    for (int bt = 0; bt < R_TILES; ++bt)
#pragma unroll
      for (int nt = 0; nt < 4; ++nt)
#pragma unroll
        for (int r = 0; r < 8; ++r) {
          float s = acc[bt][nt][r] +
                    lp[((k * R_TILES + bt) * 4 + nt) * 256 + (r + rofs) * 16 + row];
          out[(bbase + bt * 16 + r + rofs) * 256 + 64 + (nt * 16 + row) * 3 + k] = s;
        }
  }
}

// ---------------------------------------------------------------------------
extern "C" void kernel_launch(void* const* d_in, const int* in_sizes, int n_in,
                              void* d_out, int out_size, void* d_ws, size_t ws_size,
                              hipStream_t stream) {
  const float* x    = (const float*)d_in[0];
  const float* w000 = (const float*)d_in[1];
  const float* w110 = (const float*)d_in[2];
  const float* w011 = (const float*)d_in[3];
  const float* w101 = (const float*)d_in[4];
  bf16* wtiled = (bf16*)((char*)d_ws + WT_OFF);
  bf16* x0bf   = (bf16*)((char*)d_ws + X0_OFF);
  bf16* x1bf   = (bf16*)((char*)d_ws + X1_OFF);

  prep_w<<<(4 * TENSOR_ELEMS) / 256, 256, 0, stream>>>(w000, w110, w011, w101, wtiled);
  prep_x<<<(BATCH * 256) / 256, 256, 0, stream>>>(x, x0bf, x1bf);
  tp_main<<<BATCH / (R_TILES * 16), WG_THREADS, LDS_BYTES, stream>>>(wtiled, x0bf, x1bf,
                                                                     (float*)d_out);
}
